// SNN_51840255263036
// MI455X (gfx1250) — matrix operand order repaired and checked by reading, not compiled, not run on hardware
//
#include <hip/hip_runtime.h>
#include <hip/hip_bf16.h>

typedef __attribute__((ext_vector_type(16))) _Float16 v16h;
typedef __attribute__((ext_vector_type(8)))  _Float16 v8h;
typedef __attribute__((ext_vector_type(8)))  float    v8f;
typedef __attribute__((ext_vector_type(4)))  int      v4i;

#define LDS_AS __attribute__((address_space(3)))
#define GLB_AS __attribute__((address_space(1)))

#if defined(__gfx1250__)
#if __has_builtin(__builtin_amdgcn_global_load_async_to_lds_b128)
#define SNN_HAS_ASYNC 1
#else
#define SNN_HAS_ASYNC 0
#endif
#else
#define SNN_HAS_ASYNC 0
#endif

#if SNN_HAS_ASYNC
#define SNN_ASYNC_COPY16(gsrc, ldst)                                          \
  __builtin_amdgcn_global_load_async_to_lds_b128(                             \
      (GLB_AS v4i*)(GLB_AS void*)(void*)(gsrc),                               \
      (LDS_AS v4i*)(LDS_AS void*)(void*)(ldst), 0, 0)
#if __has_builtin(__builtin_amdgcn_s_wait_asynccnt)
#define SNN_WAIT_ASYNC(n) __builtin_amdgcn_s_wait_asynccnt(n)
#else
#define SNN_WAIT_ASYNC(n) asm volatile("s_wait_asynccnt %0" ::"n"(n) : "memory")
#endif
#else
#define SNN_ASYNC_COPY16(gsrc, ldst) (*(v8h*)(ldst) = *(const v8h*)(gsrc))
#define SNN_WAIT_ASYNC(n) ((void)0)
#endif

// norse constants
constexpr float kVDecay = 0.9f;   // 1 - DT*TAU_MEM_INV
constexpr float kVJump  = 0.1f;   // DT*TAU_MEM_INV
constexpr float kIDecay = 0.8f;   // 1 - DT*TAU_SYN_INV
constexpr int   T  = 128;
constexpr int   B  = 256;
constexpr int   H  = 2048;
constexpr int   D  = 8;
constexpr int   O  = 4;
constexpr int   KREAL  = 16 + H;   // [input spikes | recurrent spikes]
constexpr int   KW     = 2112;     // padded to multiple of 64 (pad cols stay 0)
constexpr int   NSTAGE = KW / 64;  // 33 pipeline stages
constexpr int   NASYNC = 5;        // async DMAs in flight per wave per stage

// ---------------------------------------------------------------------------
// 1) LIF constant-current encoder: spikes for relu(+50x) and relu(-50x)
//    xs layout: [T][B][16] f16, feat 0..7 = pos, 8..15 = neg
// ---------------------------------------------------------------------------
__global__ void SNN_encode_kernel(const float* __restrict__ x,
                                  _Float16* __restrict__ xs) {
  int tid = blockIdx.x * blockDim.x + threadIdx.x;
  if (tid >= 2 * B * D) return;
  int sign = tid / (B * D);
  int idx  = tid % (B * D);
  int b = idx / D, d = idx % D;
  float cur = 50.0f * x[b * D + d];
  if (sign) cur = -cur;
  cur = fmaxf(cur, 0.0f);
  int feat = sign * D + d;
  float v = 0.0f;
  for (int t = 0; t < T; ++t) {
    v = kVDecay * v + kVJump * cur;
    float z = (v - 1.0f) > 0.0f ? 1.0f : 0.0f;
    xs[(t * B + b) * 16 + feat] = (_Float16)z;
    v = (1.0f - z) * v;
  }
}

// ---------------------------------------------------------------------------
// 2) Build combined f16 weight matrix: wcomb[h][0:16]=w_in, [16:2064]=w_rec,
//    rest zero. Row stride KW halves.
// ---------------------------------------------------------------------------
__global__ void SNN_wconv_kernel(const float* __restrict__ w_in,
                                 const float* __restrict__ w_rec,
                                 _Float16* __restrict__ wcomb) {
  int n = H * KW;
  for (int idx = blockIdx.x * blockDim.x + threadIdx.x; idx < n;
       idx += gridDim.x * blockDim.x) {
    int h = idx / KW, c = idx % KW;
    float val = 0.0f;
    if (c < 16)         val = w_in[h * 16 + c];
    else if (c < KREAL) val = w_rec[h * H + (c - 16)];
    wcomb[idx] = (_Float16)val;
  }
}

// ---------------------------------------------------------------------------
// 3) State init: zero v,i,vo,io; m=-inf; zero both z buffers; seed z0 with xs_0
// ---------------------------------------------------------------------------
__global__ void SNN_init_kernel(_Float16* __restrict__ z0,
                                _Float16* __restrict__ z1,
                                float* __restrict__ v, float* __restrict__ cur,
                                float* __restrict__ vo, float* __restrict__ io,
                                float* __restrict__ m,
                                const _Float16* __restrict__ xs) {
  int stride = gridDim.x * blockDim.x;
  int tid = blockIdx.x * blockDim.x + threadIdx.x;
  for (int k = tid; k < B * KW; k += stride) {
    int c = k % KW, b = k / KW;
    _Float16 val = (_Float16)0.0f;
    if (c < 16) val = xs[b * 16 + c];      // t = 0 input spikes
    z0[k] = val;
    z1[k] = (_Float16)0.0f;
  }
  for (int k = tid; k < B * H; k += stride) { v[k] = 0.0f; cur[k] = 0.0f; }
  for (int k = tid; k < B * O; k += stride) {
    vo[k] = 0.0f; io[k] = 0.0f; m[k] = -3.4e38f;
  }
}

// ---------------------------------------------------------------------------
// 4) One LIF timestep: i_new = 0.8*i + [xs_t | z_{t-1}] @ [w_in | w_rec]^T,
//    fused LIF update, spikes written to z_next.
//    Grid (H/128, B/32), 256 threads = 8 waves. Wave (wy,wx) owns two 16x16
//    tiles: (b0+wy*16, h0+wx*16) and (b0+wy*16, h0+64+wx*16).
//    K pipelined in 64-wide stages through double-buffered LDS fed by
//    GLOBAL_LOAD_ASYNC_TO_LDS_B128 (ASYNCcnt), 4 WMMAs per barrier pair.
// ---------------------------------------------------------------------------
__global__ void __launch_bounds__(256)
SNN_step_kernel(const _Float16* __restrict__ zin,
                _Float16* __restrict__ zout,
                const _Float16* __restrict__ wcomb,
                float* __restrict__ v, float* __restrict__ cur,
                const _Float16* __restrict__ xs, int t) {
  __shared__ _Float16 As[2][32][80];    // 80-half row stride keeps 32B align
  __shared__ _Float16 Bs[2][128][80];
  const int tid  = threadIdx.x;
  const int lane = tid & 31;
  const int wave = tid >> 5;
  const int wy = wave >> 2;              // 0..1  batch sub-tile
  const int wx = wave & 3;               // 0..3  hidden sub-tile
  const int h0 = blockIdx.x * 128;
  const int b0 = blockIdx.y * 32;

  // staging: thread loads A: 1 x 16B ; B: 4 x 16B per stage
  const int sr = tid >> 3;               // 0..31
  const int sc = (tid & 7) * 8;          // 0,8,..,56 (halves)
  const _Float16* gA = &zin[(size_t)(b0 + sr) * KW + sc];
  const _Float16* gB = &wcomb[(size_t)(h0 + sr) * KW + sc];

  auto issue = [&](int s, int buf) {
    const int koff = s * 64;
    SNN_ASYNC_COPY16(gA + koff, &As[buf][sr][sc]);
#pragma unroll
    for (int i = 0; i < 4; ++i) {
      SNN_ASYNC_COPY16(gB + koff + (size_t)(32 * i) * KW,
                       &Bs[buf][sr + 32 * i][sc]);
    }
  };

  // fragment indices: contiguous K order (k = 16*(lane/16) + s), same for A and B
  const int arow  = wy * 16 + (lane & 15);
  const int abase = (lane >> 4) * 16;    // K 0..15 / 16..31
  const int brow  = wx * 16 + (lane & 15);
  const int bbase = (lane >> 4) * 16;    // K 0..15 / 16..31

  v8f acc0 = {};
  v8f acc1 = {};
  issue(0, 0);
  for (int s = 0; s < NSTAGE; ++s) {
    const int buf = s & 1;
    if (s + 1 < NSTAGE) {
      issue(s + 1, buf ^ 1);
      SNN_WAIT_ASYNC(NASYNC);            // stage-s DMAs done, next 5 in flight
    } else {
      SNN_WAIT_ASYNC(0);
    }
    __syncthreads();
#pragma unroll
    for (int ks = 0; ks < 2; ++ks) {
      const int kb = ks * 32;
      v8h alo = *(const v8h*)&As[buf][arow][kb + abase];
      v8h ahi = *(const v8h*)&As[buf][arow][kb + abase + 8];
      v16h a;
#pragma unroll
      for (int q = 0; q < 8; ++q) { a[q] = alo[q]; a[q + 8] = ahi[q]; }
      v16h bf0 = *(const v16h*)&Bs[buf][brow][kb + bbase];
      v16h bf1 = *(const v16h*)&Bs[buf][64 + brow][kb + bbase];
      acc0 = __builtin_amdgcn_wmma_f32_16x16x32_f16(false, a, false, bf0,
                                                    (short)0, acc0, false, false);
      acc1 = __builtin_amdgcn_wmma_f32_16x16x32_f16(false, a, false, bf1,
                                                    (short)0, acc1, false, false);
    }
    __syncthreads();
  }

  // fused LIF epilogue: C/D layout -> (b,h) element per lane/VGPR
  auto lif = [&](int bm, int hn, float a) {
    int idx = bm * H + hn;
    float vold = v[idx];
    float iold = cur[idx];
    float vdec = kVDecay * vold + kVJump * iold;
    float z = (vdec - 1.0f) > 0.0f ? 1.0f : 0.0f;
    v[idx]   = (1.0f - z) * vdec;
    cur[idx] = kIDecay * iold + a;
    zout[bm * KW + 16 + hn] = (_Float16)z;
  };
#pragma unroll
  for (int r = 0; r < 8; ++r) {
    int bm = b0 + wy * 16 + r + 8 * (lane >> 4);
    int hn = h0 + wx * 16 + (lane & 15);
    lif(bm, hn, acc0[r]);
    lif(bm, hn + 64, acc1[r]);
  }

  // hblock 0 refreshes the 16 input-spike columns for step t+1
  if (blockIdx.x == 0) {
    for (int k2 = tid; k2 < 32 * 16; k2 += blockDim.x) {
      int b = b0 + (k2 >> 4);
      int f = k2 & 15;
      _Float16 val = (_Float16)0.0f;
      if (t + 1 < T) val = xs[((t + 1) * B + b) * 16 + f];
      zout[b * KW + f] = val;
    }
  }
}

// ---------------------------------------------------------------------------
// 5) LI readout per step: vo' = 0.9 vo + 0.1 io ; io' = 0.8 io + z_t @ w_out^T
//    m = max(m, vo'). One block per batch row.
// ---------------------------------------------------------------------------
__global__ void SNN_readout_kernel(const _Float16* __restrict__ z,
                                   const float* __restrict__ w_out,
                                   float* __restrict__ vo, float* __restrict__ io,
                                   float* __restrict__ m) {
  __shared__ float red[4][256];
  int b = blockIdx.x;
  int tid = threadIdx.x;
  float p[4] = {0.f, 0.f, 0.f, 0.f};
  for (int k = tid; k < H; k += 256) {
    float zv = (float)z[b * KW + 16 + k];
    if (zv != 0.0f) {
#pragma unroll
      for (int o = 0; o < 4; ++o) p[o] += w_out[o * H + k];
    }
  }
#pragma unroll
  for (int o = 0; o < 4; ++o) red[o][tid] = p[o];
  __syncthreads();
  for (int s = 128; s > 0; s >>= 1) {
    if (tid < s) {
#pragma unroll
      for (int o = 0; o < 4; ++o) red[o][tid] += red[o][tid + s];
    }
    __syncthreads();
  }
  if (tid < 4) {
    float ioo = io[b * 4 + tid], voo = vo[b * 4 + tid];
    float von = kVDecay * voo + kVJump * ioo;
    float ion = kIDecay * ioo + red[tid][0];
    vo[b * 4 + tid] = von;
    io[b * 4 + tid] = ion;
    m[b * 4 + tid] = fmaxf(m[b * 4 + tid], von);
  }
}

// ---------------------------------------------------------------------------
// 6) softmax over the 4 outputs
// ---------------------------------------------------------------------------
__global__ void SNN_softmax_kernel(const float* __restrict__ m,
                                   float* __restrict__ out) {
  int b = blockIdx.x * blockDim.x + threadIdx.x;
  if (b >= B) return;
  float v0 = m[b*4], v1 = m[b*4+1], v2 = m[b*4+2], v3 = m[b*4+3];
  float mx = fmaxf(fmaxf(v0, v1), fmaxf(v2, v3));
  float e0 = __expf(v0 - mx), e1 = __expf(v1 - mx);
  float e2 = __expf(v2 - mx), e3 = __expf(v3 - mx);
  float s = e0 + e1 + e2 + e3;
  out[b*4] = e0/s; out[b*4+1] = e1/s; out[b*4+2] = e2/s; out[b*4+3] = e3/s;
}

// ---------------------------------------------------------------------------
extern "C" void kernel_launch(void* const* d_in, const int* in_sizes, int n_in,
                              void* d_out, int out_size, void* d_ws, size_t ws_size,
                              hipStream_t stream) {
  const float* x     = (const float*)d_in[0];
  const float* w_in  = (const float*)d_in[1];
  const float* w_rec = (const float*)d_in[2];
  const float* w_out = (const float*)d_in[3];
  float* out = (float*)d_out;

  char* p = (char*)d_ws;
  auto alloc = [&](size_t bytes) {
    char* r = p;
    p += (bytes + 255) & ~(size_t)255;
    return r;
  };
  _Float16* wcomb = (_Float16*)alloc((size_t)H * KW * 2);   // 8.7 MB
  _Float16* xs    = (_Float16*)alloc((size_t)T * B * 16 * 2);
  _Float16* z0    = (_Float16*)alloc((size_t)B * KW * 2);
  _Float16* z1    = (_Float16*)alloc((size_t)B * KW * 2);
  float* v   = (float*)alloc((size_t)B * H * 4);
  float* cur = (float*)alloc((size_t)B * H * 4);
  float* vo  = (float*)alloc((size_t)B * O * 4);
  float* io  = (float*)alloc((size_t)B * O * 4);
  float* m   = (float*)alloc((size_t)B * O * 4);

  SNN_encode_kernel<<<(2 * B * D + 255) / 256, 256, 0, stream>>>(x, xs);
  SNN_wconv_kernel<<<512, 256, 0, stream>>>(w_in, w_rec, wcomb);
  SNN_init_kernel<<<512, 256, 0, stream>>>(z0, z1, v, cur, vo, io, m, xs);

  _Float16* zc = z0;
  _Float16* zn = z1;
  for (int t = 0; t < T; ++t) {
    SNN_step_kernel<<<dim3(H / 128, B / 32), 256, 0, stream>>>(zc, zn, wcomb, v,
                                                               cur, xs, t);
    SNN_readout_kernel<<<B, 256, 0, stream>>>(zn, w_out, vo, io, m);
    _Float16* tmp = zc; zc = zn; zn = tmp;
  }
  SNN_softmax_kernel<<<1, 256, 0, stream>>>(m, out);
}
